// RoutesEncoder_71940702208695
// MI455X (gfx1250) — matrix operand order repaired and checked by reading, not compiled, not run on hardware
//
#include <hip/hip_runtime.h>
#include <hip/hip_bf16.h>

typedef __attribute__((ext_vector_type(16))) __bf16 v16bf;
typedef __attribute__((ext_vector_type(8)))  float  v8f;
typedef unsigned int v4u __attribute__((ext_vector_type(4)));
typedef int          v8i __attribute__((ext_vector_type(8)));
typedef int          v4i __attribute__((ext_vector_type(4)));

// ---------- bf16 split helpers (bit-level, buffers are u16 bit patterns) ----
__device__ __forceinline__ unsigned short f2bf_bits(float f) {
    unsigned u = __float_as_uint(f);
    unsigned r = u + 0x7FFFu + ((u >> 16) & 1u);   // round-to-nearest-even
    return (unsigned short)(r >> 16);
}
__device__ __forceinline__ float bf_bits2f(unsigned short s) {
    return __uint_as_float(((unsigned)s) << 16);
}

// ---------------------------------------------------------------------------
// Kernel 1: per-weight transpose + hi/lo bf16 split.  W[K,N] f32 -> Wt[N,K].
// ---------------------------------------------------------------------------
__global__ void wsplit_kernel(const float* __restrict__ W,
                              unsigned short* __restrict__ Th,
                              unsigned short* __restrict__ Tl,
                              int K, int N) {
    int o = blockIdx.x * blockDim.x + threadIdx.x;       // index into Wt
    if (o >= K * N) return;
    int n = o / K;
    int k = o - n * K;
    float v  = W[(size_t)k * N + n];
    unsigned short hb = f2bf_bits(v);
    Th[o] = hb;
    Tl[o] = f2bf_bits(v - bf_bits2f(hb));
}

// ---------------------------------------------------------------------------
// Kernel 2: gather + max over route positions, hi/lo split.
// ---------------------------------------------------------------------------
__global__ __launch_bounds__(128)
void gather_max_kernel(const float* __restrict__ G,          // [100000,512]
                       const int*   __restrict__ idx,        // [16384,32]
                       unsigned short* __restrict__ Xh,      // [16384,512]
                       unsigned short* __restrict__ Xl) {
    const int r = blockIdx.x;
    const int t = threadIdx.x;                                // 0..127 (float4 lane)
    const int* loc = idx + r * 32;
    float4 m = make_float4(-3.4e38f, -3.4e38f, -3.4e38f, -3.4e38f);
    #pragma unroll 4
    for (int j = 0; j < 32; ++j) {
        const float4* row = (const float4*)(G + (size_t)loc[j] * 512);
        float4 v = row[t];
        m.x = fmaxf(m.x, v.x); m.y = fmaxf(m.y, v.y);
        m.z = fmaxf(m.z, v.z); m.w = fmaxf(m.w, v.w);
    }
    const size_t base = (size_t)r * 512 + (size_t)t * 4;
    float vv[4] = {m.x, m.y, m.z, m.w};
    #pragma unroll
    for (int c = 0; c < 4; ++c) {
        unsigned short hb = f2bf_bits(vv[c]);
        Xh[base + c] = hb;
        Xl[base + c] = f2bf_bits(vv[c] - bf_bits2f(hb));
    }
}

// ---------------------------------------------------------------------------
// TDM descriptor: 2-D bf16 tile load  (tile = 32 K-elems x 64 N-rows)
//   tensor: Bt[N, K] row-major, element = 2 bytes (data_size code 1)
//   tile start = Bt + (nBase*K + kk) elems ; dest = LDS byte offset lds_off
// D# per ISA 8.3/8.4: group0 {count=1, lds_addr, global_addr, type=2},
// group1 {data_size, tensor_dim0=K, tensor_dim1=N, tile_dim0=32, tile_dim1=64,
//         tensor_dim0_stride=K}
// ---------------------------------------------------------------------------
__device__ __forceinline__ void tdm_load_tile(const unsigned short* base,
                                              unsigned lds_off,
                                              int nBase, int kk, int K, int N) {
    unsigned long long g = (unsigned long long)(size_t)base +
                           ((unsigned long long)nBase * (unsigned)K + (unsigned)kk) * 2ull;
    v4u g0;
    g0[0] = 1u;                                           // count=1, user D#
    g0[1] = lds_off;                                      // lds_addr (bytes)
    g0[2] = (unsigned)(g & 0xFFFFFFFFull);                // global_addr[31:0]
    g0[3] = (unsigned)((g >> 32) & 0x1FFFFFFull) | (2u << 30); // addr[56:32] | type=2

    v8i g1;
    g1[0] = (int)(1u << 16);                              // data_size = 2 bytes
    g1[1] = (int)(((unsigned)K & 0xFFFFu) << 16);         // tensor_dim0 lo16
    g1[2] = (int)((((unsigned)K >> 16) & 0xFFFFu) |
                  (((unsigned)N & 0xFFFFu) << 16));       // dim0 hi | dim1 lo
    g1[3] = (int)((((unsigned)N >> 16) & 0xFFFFu) |
                  (32u << 16));                           // dim1 hi | tile_dim0=32
    g1[4] = (int)64u;                                     // tile_dim1=64, tile_dim2=0
    g1[5] = (int)(unsigned)K;                             // tensor_dim0_stride lo32
    g1[6] = 0;                                            // stride hi | dim1_stride lo
    g1[7] = 0;

    v4i g2 = {0, 0, 0, 0};
    v4i g3 = {0, 0, 0, 0};
    v8i g4 = {0, 0, 0, 0, 0, 0, 0, 0};
    __builtin_amdgcn_tensor_load_to_lds(g0, g1, g2, g3, g4, 0);
}

// ---------------------------------------------------------------------------
// Kernel 3: split-bf16 WMMA GEMM + bias + ReLU, TDM-staged B tiles in LDS.
//   C[M,N] = relu(A[M,K] * Bt[N,K]^T + bias)
// Block = 8 waves = 128(M) x 64(N); wave tile = 16(M) x 64(N).
// B tile (64 x 32, hi+lo) double-buffered in LDS via tensor_load_to_lds.
// ---------------------------------------------------------------------------
__global__ __launch_bounds__(256)
void gemm_split_kernel(const unsigned short* __restrict__ Ah,
                       const unsigned short* __restrict__ Al,
                       const unsigned short* __restrict__ Bth,
                       const unsigned short* __restrict__ Btl,
                       const float* __restrict__ bias,
                       unsigned short* __restrict__ Oh,
                       unsigned short* __restrict__ Ol,
                       float* __restrict__ Of,
                       int M, int N, int K) {
    __shared__ alignas(16) unsigned short ldsB[2][2][64 * 32]; // [buf][hi/lo] 4KB each

    const int lane   = threadIdx.x & 31;
    const int waveId = threadIdx.x >> 5;
    const int nQuads = N >> 6;
    const int mGroup = blockIdx.x / nQuads;
    const int nQuad  = blockIdx.x - mGroup * nQuads;
    const int mTile  = mGroup * 8 + waveId;

    const int lo4 = lane & 15;
    const int kh  = lane >> 4;                      // K-half select per ISA layout
    const int mRow  = mTile * 16 + lo4;
    const int nBase = nQuad * 64;

    const unsigned short* arh = Ah + (size_t)mRow * K;
    const unsigned short* arl = Al + (size_t)mRow * K;

    v8f acc[4];
    #pragma unroll
    for (int t = 0; t < 4; ++t)
        #pragma unroll
        for (int i = 0; i < 8; ++i) acc[t][i] = 0.0f;

    typedef union { uint4 q[2]; v16bf v; } frag_t;

    // LDS byte offsets of the four staging buffers (flat addr low 32 bits)
    const unsigned ldsBase = (unsigned)(size_t)(&ldsB[0][0][0]);
    const unsigned bufOff[2][2] = {
        { ldsBase + 0u * 4096u, ldsBase + 1u * 4096u },
        { ldsBase + 2u * 4096u, ldsBase + 3u * 4096u } };

    // uniform (scalar-branch) TDM issuer: wave 0 of the block
    const bool tdmWave = (__builtin_amdgcn_readfirstlane(waveId) == 0);

    if (tdmWave) {
        tdm_load_tile(Bth, bufOff[0][0], nBase, 0, K, N);
        tdm_load_tile(Btl, bufOff[0][1], nBase, 0, K, N);
        __builtin_amdgcn_s_wait_tensorcnt(0);
    }
    __syncthreads();

    int cur = 0;
    for (int k = 0; k < K; k += 32) {
        // prefetch next K-slab into the other buffer (overlaps compute)
        if (tdmWave && (k + 32 < K)) {
            tdm_load_tile(Bth, bufOff[cur ^ 1][0], nBase, k + 32, K, N);
            tdm_load_tile(Btl, bufOff[cur ^ 1][1], nBase, k + 32, K, N);
        }

        // A fragment (16x32 bf16) from global: two 16B chunks per lane
        frag_t fah, fal;
        {
            const uint4* p = (const uint4*)(arh + k + kh * 8);
            fah.q[0] = p[0]; fah.q[1] = p[2];
            const uint4* q = (const uint4*)(arl + k + kh * 8);
            fal.q[0] = q[0]; fal.q[1] = q[2];
        }

        #pragma unroll
        for (int t = 0; t < 4; ++t) {
            // B fragment from LDS tile: row n_local, two 16B chunks at k 8*kh and 16+8*kh
            const int rowOff = (t * 16 + lo4) * 32 + kh * 8;
            frag_t fbh, fbl;
            const uint4* pb = (const uint4*)(&ldsB[cur][0][rowOff]);
            fbh.q[0] = pb[0]; fbh.q[1] = pb[2];
            const uint4* qb = (const uint4*)(&ldsB[cur][1][rowOff]);
            fbl.q[0] = qb[0]; fbl.q[1] = qb[2];

            acc[t] = __builtin_amdgcn_wmma_f32_16x16x32_bf16(
                         false, fah.v, false, fbh.v, (short)0, acc[t], false, false);
            acc[t] = __builtin_amdgcn_wmma_f32_16x16x32_bf16(
                         false, fah.v, false, fbl.v, (short)0, acc[t], false, false);
            acc[t] = __builtin_amdgcn_wmma_f32_16x16x32_bf16(
                         false, fal.v, false, fbh.v, (short)0, acc[t], false, false);
        }

        if (tdmWave) __builtin_amdgcn_s_wait_tensorcnt(0);
        __syncthreads();
        cur ^= 1;
    }

    // Epilogue: C/D layout — lane holds column n = lane&15, rows 8*kh + v.
    #pragma unroll
    for (int t = 0; t < 4; ++t) {
        const int n = nBase + t * 16 + lo4;
        const float b = bias[n];
        #pragma unroll
        for (int v = 0; v < 8; ++v) {
            const int m = mTile * 16 + kh * 8 + v;
            float val = fmaxf(acc[t][v] + b, 0.0f);
            const size_t off = (size_t)m * N + n;
            if (Of) {
                Of[off] = val;
            } else {
                unsigned short hb = f2bf_bits(val);
                Oh[off] = hb;
                Ol[off] = f2bf_bits(val - bf_bits2f(hb));
            }
        }
    }
}

// ---------------------------------------------------------------------------
extern "C" void kernel_launch(void* const* d_in, const int* in_sizes, int n_in,
                              void* d_out, int out_size, void* d_ws, size_t ws_size,
                              hipStream_t stream) {
    const int N_ROUTES = 16384, D_GRAPH = 512, D_ROUTE = 1024;

    const float* G    = (const float*)d_in[0];
    const int*   idx  = (const int*)  d_in[1];
    const float* W0   = (const float*)d_in[2];
    const float* b0   = (const float*)d_in[3];
    const float* W1   = (const float*)d_in[4];
    const float* b1   = (const float*)d_in[5];
    const float* W2   = (const float*)d_in[6];
    const float* b2   = (const float*)d_in[7];
    float* out        = (float*)d_out;

    // Workspace layout (u16 bf16-bit buffers)
    const size_t SZ_X0 = (size_t)N_ROUTES * D_GRAPH;   // elems
    const size_t SZ_X1 = (size_t)N_ROUTES * D_ROUTE;
    const size_t SZ_W0 = (size_t)D_GRAPH * D_ROUTE;
    const size_t SZ_W1 = (size_t)D_ROUTE * D_ROUTE;

    unsigned short* p = (unsigned short*)d_ws;
    unsigned short* X0h = p; p += SZ_X0;
    unsigned short* X0l = p; p += SZ_X0;
    unsigned short* X1h = p; p += SZ_X1;
    unsigned short* X1l = p; p += SZ_X1;
    unsigned short* X2h = p; p += SZ_X1;
    unsigned short* X2l = p; p += SZ_X1;
    unsigned short* T0h = p; p += SZ_W0;
    unsigned short* T0l = p; p += SZ_W0;
    unsigned short* T1h = p; p += SZ_W1;
    unsigned short* T1l = p; p += SZ_W1;
    unsigned short* T2h = p; p += SZ_W1;
    unsigned short* T2l = p; p += SZ_W1;

    // 1) weight transpose + split
    {
        int n0 = (int)SZ_W0, n1 = (int)SZ_W1;
        wsplit_kernel<<<(n0 + 255) / 256, 256, 0, stream>>>(W0, T0h, T0l, D_GRAPH, D_ROUTE);
        wsplit_kernel<<<(n1 + 255) / 256, 256, 0, stream>>>(W1, T1h, T1l, D_ROUTE, D_ROUTE);
        wsplit_kernel<<<(n1 + 255) / 256, 256, 0, stream>>>(W2, T2h, T2l, D_ROUTE, D_ROUTE);
    }

    // 2) gather + max
    gather_max_kernel<<<N_ROUTES, 128, 0, stream>>>(G, idx, X0h, X0l);

    // 3) GEMM chain: block = 128(M) x 64(N)
    const int M = N_ROUTES, N = D_ROUTE;
    const int blocks = (M / 128) * (N / 64);

    gemm_split_kernel<<<blocks, 256, 0, stream>>>(X0h, X0l, T0h, T0l, b0,
                                                  X1h, X1l, nullptr, M, N, D_GRAPH);
    gemm_split_kernel<<<blocks, 256, 0, stream>>>(X1h, X1l, T1h, T1l, b1,
                                                  X2h, X2l, nullptr, M, N, D_ROUTE);
    gemm_split_kernel<<<blocks, 256, 0, stream>>>(X2h, X2l, T2h, T2l, b2,
                                                  nullptr, nullptr, out, M, N, D_ROUTE);
}
